// PhysicsGraphTransformer_38010460570143
// MI455X (gfx1250) — matrix operand order repaired and checked by reading, not compiled, not run on hardware
//
#include <hip/hip_runtime.h>
#include <math.h>

// ---------------------------------------------------------------------------
// Problem constants (match reference)
// ---------------------------------------------------------------------------
static constexpr int N    = 20000;
static constexpr int E    = 320000;
static constexpr int D    = 128;
static constexpr int H    = 8;
static constexpr int L    = 4;
static constexpr int DFF  = 512;
static constexpr int DH   = 16;
static constexpr int EFW  = 2 * D + D / 2;   // 320
static constexpr int FEAT = D / 2;           // 64
static constexpr int HID1 = D / 4;           // 32
static constexpr int NEP  = 4;
static constexpr int CH   = 65536;           // edge chunk for head MLPs
#define SCALE_QK 0.25f                        // 1/sqrt(16)

typedef __attribute__((ext_vector_type(16))) _Float16 v16h;
typedef __attribute__((ext_vector_type(8)))  float    v8f;
typedef __attribute__((ext_vector_type(4)))  int      v4i;
typedef __attribute__((ext_vector_type(4)))  unsigned v4u;
typedef __attribute__((ext_vector_type(8)))  unsigned v8u;

union HF16 { v4i q[2]; v16h v; };

// ---------------------------------------------------------------------------
// Tensor Data Mover: 1D bulk copy global -> LDS (contiguous nbytes, 8B units).
// Builds D# group0 (count/lds_addr/global_addr/type=2) and group1
// (data_size=8B, tensor_dim0=tile_dim0=n8, tensor_dim1=tile_dim1=1,
//  tensor_dim0_stride=n8) per CDNA5 ISA 8.3/8.4, issues TENSOR_LOAD_TO_LDS.
// All inputs must be wave-uniform. Tracked by TENSORcnt.
// ---------------------------------------------------------------------------
__device__ __forceinline__ void tdm_load_1d(unsigned lds_off, const void* gptr,
                                            unsigned nbytes) {
  unsigned long long ga = (unsigned long long)(uintptr_t)gptr;
  unsigned n8 = nbytes >> 3;
  v4u g0;
  g0[0] = 1u;                                              // count=1, user mode
  g0[1] = lds_off;                                         // lds_addr
  g0[2] = (unsigned)ga;                                    // global_addr[31:0]
  g0[3] = (unsigned)((ga >> 32) & 0x01FFFFFFu) | (2u << 30); // addr[56:32]|type=2
  v8u g1;
  g1[0] = 3u << 16;                  // workgroup_mask=0, data_size=3 (8 bytes)
  g1[1] = (n8 & 0xFFFFu) << 16;      // tensor_dim0[15:0]
  g1[2] = ((n8 >> 16) & 0xFFFFu) | (1u << 16);  // tensor_dim0[31:16] | tensor_dim1=1
  g1[3] = (n8 & 0xFFFFu) << 16;      // tile_dim0 = n8
  g1[4] = 1u;                        // tile_dim1=1, tile_dim2=0
  g1[5] = n8;                        // tensor_dim0_stride[31:0]
  g1[6] = 0u;                        // stride hi / dim1_stride lo
  g1[7] = 0u;
  asm volatile("tensor_load_to_lds %0, %1" :: "s"(g0), "s"(g1) : "memory");
}

__device__ __forceinline__ unsigned lds_off_of(const void* p) {
  return (unsigned)(uintptr_t)p;     // flat LDS addr: offset lives in low 32b
}

// ---------------------------------------------------------------------------
// Weight repack: f32 [K,Nc] row-major -> f16 WMMA B fragments.
//   lane L holds column n=(L&15), k = ktile*32 + (L>>4)*16 + i  (i=0..15)
//   packed offset = (((ntile*(K/32) + ktile)*32) + lane)*16 + i
// ---------------------------------------------------------------------------
__global__ void k_pack_b(const float* __restrict__ B, _Float16* __restrict__ out,
                         int K, int Nc) {
  int idx = blockIdx.x * blockDim.x + threadIdx.x;
  if (idx >= K * Nc) return;
  int i    = idx & 15;
  int lane = (idx >> 4) & 31;
  int rest = idx >> 9;
  int ks   = K >> 5;
  int s    = rest % ks;
  int t    = rest / ks;
  int k    = s * 32 + (lane >> 4) * 16 + i;
  int n    = t * 16 + (lane & 15);
  out[idx] = (_Float16)B[(size_t)k * Nc + n];
}

// ---------------------------------------------------------------------------
// WMMA GEMM: C[M,Nc] = act(A_f16[M,K] @ Bpacked + bias)
// 256-thread blocks = 8 waves; block tile 128 rows x 64 cols (NT=4).
// Packed B for the block's 4 N-tiles (contiguous span) is DMA'd to LDS by the
// TDM once per block and shared by all 8 waves. A fragments double-buffered.
// grid = (ceil(M/128), Nc/64), dynamic LDS = NT*(K/32)*1024 bytes.
// ---------------------------------------------------------------------------
template <int NT, bool RELU>
__global__ __launch_bounds__(256)
void k_gemm16(const _Float16* __restrict__ A, const _Float16* __restrict__ Bp,
              const float* __restrict__ bias, float* __restrict__ Cf,
              _Float16* __restrict__ Ch, int M, int K, int Nc) {
  extern __shared__ _Float16 lds_b[];
  const int tid  = threadIdx.x;
  const int lane = tid & 31;
  const int wid  = tid >> 5;
  const int mr = lane & 15, hs = lane >> 4;
  const int m0 = (blockIdx.x * 8 + wid) * 16;
  const int t0 = blockIdx.y * NT;
  const int ks = K >> 5;

  // one TDM per block stages NT*ks*512 halves of packed B into LDS
  if (__builtin_amdgcn_readfirstlane(tid) < 32) {
    tdm_load_1d(lds_off_of(lds_b), Bp + (size_t)t0 * ks * 512,
                (unsigned)(NT * ks * 1024));
    __builtin_amdgcn_s_wait_tensorcnt(0);
  }
  __syncthreads();

  int arowi = m0 + mr;
  if (arowi > M - 1) arowi = M - 1;
  const _Float16* arow = A + (size_t)arowi * K;

  v8f acc[NT];
#pragma unroll
  for (int t = 0; t < NT; ++t) {
    float bv = bias ? bias[(t0 + t) * 16 + mr] : 0.0f;
    v8f z = {bv, bv, bv, bv, bv, bv, bv, bv};
    acc[t] = z;
  }

  HF16 a_cur, a_nxt;
  a_cur.q[0] = *(const v4i*)(arow + hs * 16);
  a_cur.q[1] = *(const v4i*)(arow + hs * 16 + 8);
  for (int s = 0; s < ks; ++s) {
    int sn = (s + 1 < ks) ? s + 1 : s;
    a_nxt.q[0] = *(const v4i*)(arow + sn * 32 + hs * 16);
    a_nxt.q[1] = *(const v4i*)(arow + sn * 32 + hs * 16 + 8);
    HF16 b[NT];
#pragma unroll
    for (int t = 0; t < NT; ++t) {
      const _Float16* lb = lds_b + ((size_t)(t * ks + s) * 32 + lane) * 16;
      b[t].q[0] = *(const v4i*)lb;
      b[t].q[1] = *(const v4i*)(lb + 8);
    }
#pragma unroll
    for (int t = 0; t < NT; ++t)
      acc[t] = __builtin_amdgcn_wmma_f32_16x16x32_f16(
          false, a_cur.v, false, b[t].v, (short)0, acc[t], false, false);
    a_cur = a_nxt;
  }

#pragma unroll
  for (int t = 0; t < NT; ++t) {
    int n = (t0 + t) * 16 + mr;
#pragma unroll
    for (int v = 0; v < 8; ++v) {
      int m = m0 + v + 8 * hs;
      if (m >= M) continue;
      float val = acc[t][v];
      if (RELU) val = fmaxf(val, 0.0f);
      if (Cf) Cf[(size_t)m * Nc + n] = val;
      if (Ch) Ch[(size_t)m * Nc + n] = (_Float16)val;
    }
  }
}

// ---------------------------------------------------------------------------
// Gather-GEMM for edge heads: A row e = [h16[src[e]](128) | h16[dst[e]](128)
// | efeat16[e](64)], K = 320 (10 k-tiles). Same TDM-staged B, 8 waves/block.
// Output f16 chunk-local [Ecnt, Nc], relu'd.
// ---------------------------------------------------------------------------
__global__ __launch_bounds__(256)
void k_gemm_ef(const _Float16* __restrict__ h16, const _Float16* __restrict__ ef16,
               const int* __restrict__ src, const int* __restrict__ dst,
               const _Float16* __restrict__ Bp, const float* __restrict__ bias,
               _Float16* __restrict__ Ch, int e0, int Ecnt, int Nc) {
  extern __shared__ _Float16 lds_b[];
  constexpr int NT = 4;
  constexpr int ks = EFW / 32;  // 10
  const int tid  = threadIdx.x;
  const int lane = tid & 31;
  const int wid  = tid >> 5;
  const int mr = lane & 15, hs = lane >> 4;
  const int ml = (blockIdx.x * 8 + wid) * 16;
  const int t0 = blockIdx.y * NT;

  if (__builtin_amdgcn_readfirstlane(tid) < 32) {
    tdm_load_1d(lds_off_of(lds_b), Bp + (size_t)t0 * ks * 512,
                (unsigned)(NT * ks * 1024));
    __builtin_amdgcn_s_wait_tensorcnt(0);
  }
  __syncthreads();

  int eg = e0 + ml + mr;
  int emax = e0 + Ecnt - 1;
  if (eg > emax) eg = emax;
  const int sE = src[eg], dE = dst[eg];
  const _Float16* ra = h16 + (size_t)sE * D;
  const _Float16* rb = h16 + (size_t)dE * D;
  const _Float16* rc = ef16 + (size_t)eg * FEAT;

  auto aload = [&](int s, HF16& a) {
    int k0 = s * 32;
    const _Float16* rowp;
    int koff;
    if (k0 < 128)      { rowp = ra; koff = k0; }
    else if (k0 < 256) { rowp = rb; koff = k0 - 128; }
    else               { rowp = rc; koff = k0 - 256; }
    a.q[0] = *(const v4i*)(rowp + koff + hs * 16);
    a.q[1] = *(const v4i*)(rowp + koff + hs * 16 + 8);
  };

  v8f acc[NT];
#pragma unroll
  for (int t = 0; t < NT; ++t) {
    float bv = bias[(t0 + t) * 16 + mr];
    v8f z = {bv, bv, bv, bv, bv, bv, bv, bv};
    acc[t] = z;
  }

  HF16 a_cur, a_nxt;
  aload(0, a_cur);
  for (int s = 0; s < ks; ++s) {
    aload((s + 1 < ks) ? s + 1 : s, a_nxt);
    HF16 b[NT];
#pragma unroll
    for (int t = 0; t < NT; ++t) {
      const _Float16* lb = lds_b + ((size_t)(t * ks + s) * 32 + lane) * 16;
      b[t].q[0] = *(const v4i*)lb;
      b[t].q[1] = *(const v4i*)(lb + 8);
    }
#pragma unroll
    for (int t = 0; t < NT; ++t)
      acc[t] = __builtin_amdgcn_wmma_f32_16x16x32_f16(
          false, a_cur.v, false, b[t].v, (short)0, acc[t], false, false);
    a_cur = a_nxt;
  }

#pragma unroll
  for (int t = 0; t < NT; ++t) {
    int n = (t0 + t) * 16 + mr;
#pragma unroll
    for (int v = 0; v < 8; ++v) {
      int m = ml + v + 8 * hs;
      if (m >= Ecnt) continue;
      float val = fmaxf(acc[t][v], 0.0f);
      Ch[(size_t)m * Nc + n] = (_Float16)val;
    }
  }
}

// ---------------------------------------------------------------------------
// h = x @ in_w + in_b   (K=16, pure VALU; negligible cost)
// ---------------------------------------------------------------------------
__global__ void k_node_proj(const float* __restrict__ x, const float* __restrict__ w,
                            const float* __restrict__ b, float* __restrict__ h,
                            _Float16* __restrict__ h16) {
  int idx = blockIdx.x * blockDim.x + threadIdx.x;
  if (idx >= N * D) return;
  int node = idx >> 7, c = idx & 127;
  const float* xr = x + (size_t)node * 16;
  float acc = b[c];
#pragma unroll
  for (int j = 0; j < 16; ++j) acc = fmaf(xr[j], w[j * D + c], acc);
  h[idx] = acc;
  h16[idx] = (_Float16)acc;
}

// ---------------------------------------------------------------------------
// Edge embedding MLP: one wave per edge; shfl broadcast of 32 hidden units.
// ---------------------------------------------------------------------------
__global__ __launch_bounds__(256)
void k_edge_embed(const float* __restrict__ ea, const float* __restrict__ w1,
                  const float* __restrict__ b1, const float* __restrict__ w2,
                  const float* __restrict__ b2, _Float16* __restrict__ efeat,
                  float* __restrict__ ew) {
  int e = blockIdx.x * 8 + (threadIdx.x >> 5);
  if (e >= E) return;
  int lane = threadIdx.x & 31;
  float a = ea[e];
  float hj = fmaxf(fmaf(a, w1[lane], b1[lane]), 0.0f);
  float f0 = b2[lane], f1 = b2[lane + 32];
#pragma unroll
  for (int j = 0; j < HID1; ++j) {
    float hjj = __shfl(hj, j, 32);
    f0 = fmaf(hjj, w2[j * FEAT + lane], f0);
    f1 = fmaf(hjj, w2[j * FEAT + lane + 32], f1);
  }
  efeat[(size_t)e * FEAT + lane] = (_Float16)f0;
  efeat[(size_t)e * FEAT + lane + 32] = (_Float16)f1;
  float t = f0 + f1;
#pragma unroll
  for (int m = 1; m < 32; m <<= 1) t += __shfl_xor(t, m, 32);
  if (lane == 0) ew[e] = 1.0f / (1.0f + expf(-t));
}

// ---------------------------------------------------------------------------
// Attention edge scatter: one wave per edge; 4 f32 elements per lane;
// head = lane>>2; f32 atomic scatter into sums/cnt (L2-resident).
// ---------------------------------------------------------------------------
__global__ __launch_bounds__(256)
void k_attn_edge(const float* __restrict__ Q, const float* __restrict__ Kx,
                 const float* __restrict__ V, const int* __restrict__ src,
                 const int* __restrict__ dst, const float* __restrict__ ew,
                 float* __restrict__ sums, float* __restrict__ cnt) {
  int e = blockIdx.x * 8 + (threadIdx.x >> 5);
  if (e >= E) return;
  int lane = threadIdx.x & 31;
  int s = src[e], d = dst[e];
  int base = lane * 4;
  float4 q = *(const float4*)(Q + (size_t)s * D + base);
  float4 k = *(const float4*)(Kx + (size_t)d * D + base);
  float part = q.x * k.x + q.y * k.y + q.z * k.z + q.w * k.w;
  part += __shfl_xor(part, 1, 32);
  part += __shfl_xor(part, 2, 32);
  float score = part * SCALE_QK * ew[e];
  float4 v = *(const float4*)(V + (size_t)d * D + base);
  float* sp = sums + (size_t)d * D + base;
  atomicAdd(sp + 0, v.x * score);
  atomicAdd(sp + 1, v.y * score);
  atomicAdd(sp + 2, v.z * score);
  atomicAdd(sp + 3, v.w * score);
  if (lane == 0) atomicAdd(cnt + d, 1.0f);
}

// segment mean -> f16 for the Wo GEMM
__global__ void k_segmean(const float* __restrict__ sums, const float* __restrict__ cnt,
                          _Float16* __restrict__ out16) {
  int idx = blockIdx.x * blockDim.x + threadIdx.x;
  if (idx >= N * D) return;
  float c = fmaxf(cnt[idx >> 7], 1.0f);
  out16[idx] = (_Float16)(sums[idx] / c);
}

// ---------------------------------------------------------------------------
// Fused residual + LayerNorm, one wave per row; writes f32 + f16 copies.
// ---------------------------------------------------------------------------
__global__ __launch_bounds__(256)
void k_ln_res(float* __restrict__ h, _Float16* __restrict__ h16,
              const float* __restrict__ a, const float* __restrict__ s,
              const float* __restrict__ b) {
  int row = blockIdx.x * 8 + (threadIdx.x >> 5);
  if (row >= N) return;
  int lane = threadIdx.x & 31;
  size_t off = (size_t)row * D + lane * 4;
  float4 hv = *(const float4*)(h + off);
  float4 av = *(const float4*)(a + off);
  float x0 = hv.x + av.x, x1 = hv.y + av.y, x2 = hv.z + av.z, x3 = hv.w + av.w;
  float sum = x0 + x1 + x2 + x3;
#pragma unroll
  for (int m = 1; m < 32; m <<= 1) sum += __shfl_xor(sum, m, 32);
  float mu = sum * (1.0f / D);
  float d0 = x0 - mu, d1 = x1 - mu, d2 = x2 - mu, d3 = x3 - mu;
  float vs = d0 * d0 + d1 * d1 + d2 * d2 + d3 * d3;
#pragma unroll
  for (int m = 1; m < 32; m <<= 1) vs += __shfl_xor(vs, m, 32);
  float inv = rsqrtf(vs * (1.0f / D) + 1e-5f);
  int c = lane * 4;
  float y0 = d0 * inv * s[c + 0] + b[c + 0];
  float y1 = d1 * inv * s[c + 1] + b[c + 1];
  float y2 = d2 * inv * s[c + 2] + b[c + 2];
  float y3 = d3 * inv * s[c + 3] + b[c + 3];
  float4 yo = {y0, y1, y2, y3};
  *(float4*)(h + off) = yo;
  h16[off + 0] = (_Float16)y0; h16[off + 1] = (_Float16)y1;
  h16[off + 2] = (_Float16)y2; h16[off + 3] = (_Float16)y3;
}

// Final tiny head layer: per-edge dot(z2[64], w3[64,NO]) (+softplus for pp)
template <int NO, bool SP>
__global__ void k_headout(const _Float16* __restrict__ z2, const float* __restrict__ w3,
                          const float* __restrict__ b3, float* __restrict__ out,
                          int Ecnt) {
  int e = blockIdx.x * blockDim.x + threadIdx.x;
  if (e >= Ecnt) return;
  const _Float16* zr = z2 + (size_t)e * 64;
  float acc[NO];
#pragma unroll
  for (int o = 0; o < NO; ++o) acc[o] = b3[o];
  for (int j = 0; j < 64; ++j) {
    float zv = (float)zr[j];
#pragma unroll
    for (int o = 0; o < NO; ++o) acc[o] = fmaf(zv, w3[j * NO + o], acc[o]);
  }
#pragma unroll
  for (int o = 0; o < NO; ++o) {
    float r = acc[o];
    if (SP) r = fmaxf(r, 0.0f) + log1pf(expf(-fabsf(r))) + 1e-6f;  // softplus
    out[(size_t)e * NO + o] = r;
  }
}

__global__ void k_zero(float* __restrict__ p, int n) {
  int i = blockIdx.x * blockDim.x + threadIdx.x;
  if (i < n) p[i] = 0.0f;
}

// ---------------------------------------------------------------------------
// Host orchestration
// ---------------------------------------------------------------------------
extern "C" void kernel_launch(void* const* d_in, const int* in_sizes, int n_in,
                              void* d_out, int out_size, void* d_ws, size_t ws_size,
                              hipStream_t stream) {
  const float* x     = (const float*)d_in[0];
  const int*   ei    = (const int*)d_in[1];
  const float* ea    = (const float*)d_in[2];
  const float* in_w  = (const float*)d_in[3];
  const float* in_b  = (const float*)d_in[4];
  const float* ee_w1 = (const float*)d_in[5];
  const float* ee_b1 = (const float*)d_in[6];
  const float* ee_w2 = (const float*)d_in[7];
  const float* ee_b2 = (const float*)d_in[8];
  const float* Wq    = (const float*)d_in[9];
  const float* Wk    = (const float*)d_in[10];
  const float* Wv    = (const float*)d_in[11];
  const float* Wo    = (const float*)d_in[12];
  const float* Wo_b  = (const float*)d_in[13];
  const float* ff_w1 = (const float*)d_in[14];
  const float* ff_b1 = (const float*)d_in[15];
  const float* ff_w2 = (const float*)d_in[16];
  const float* ff_b2 = (const float*)d_in[17];
  const float* ln1_s = (const float*)d_in[18];
  const float* ln1_b = (const float*)d_in[19];
  const float* ln2_s = (const float*)d_in[20];
  const float* ln2_b = (const float*)d_in[21];
  const float* ec_w1 = (const float*)d_in[22];
  const float* ec_b1 = (const float*)d_in[23];
  const float* ec_w2 = (const float*)d_in[24];
  const float* ec_b2 = (const float*)d_in[25];
  const float* ec_w3 = (const float*)d_in[26];
  const float* ec_b3 = (const float*)d_in[27];
  const float* pp_w1 = (const float*)d_in[28];
  const float* pp_b1 = (const float*)d_in[29];
  const float* pp_w2 = (const float*)d_in[30];
  const float* pp_b2 = (const float*)d_in[31];
  const float* pp_w3 = (const float*)d_in[32];
  const float* pp_b3 = (const float*)d_in[33];
  const int* src = ei;
  const int* dst = ei + E;
  float* out = (float*)d_out;

  // workspace carve (256B aligned)
  char* wp = (char*)d_ws;
  auto carve = [&](size_t bytes) -> char* {
    char* r = wp;
    wp += (bytes + 255) & ~(size_t)255;
    return r;
  };
  float*    h      = (float*)carve((size_t)N * D * 4);
  _Float16* h16    = (_Float16*)carve((size_t)N * D * 2);
  float*    Qb     = (float*)carve((size_t)N * D * 4);
  float*    Kb     = (float*)carve((size_t)N * D * 4);
  float*    Vb     = (float*)carve((size_t)N * D * 4);
  float*    sums   = (float*)carve((size_t)N * D * 4);
  float*    cnt    = (float*)carve((size_t)N * 4);
  _Float16* msg16  = (_Float16*)carve((size_t)N * D * 2);
  float*    aout   = (float*)carve((size_t)N * D * 4);
  _Float16* hid16  = (_Float16*)carve((size_t)N * DFF * 2);
  float*    ew     = (float*)carve((size_t)E * 4);
  _Float16* efeat  = (_Float16*)carve((size_t)E * FEAT * 2);
  _Float16* z1     = (_Float16*)carve((size_t)CH * D * 2);
  _Float16* z2     = (_Float16*)carve((size_t)CH * 64 * 2);
  _Float16* wq_p   = (_Float16*)carve((size_t)L * D * D * 2);
  _Float16* wk_p   = (_Float16*)carve((size_t)L * D * D * 2);
  _Float16* wv_p   = (_Float16*)carve((size_t)L * D * D * 2);
  _Float16* wo_p   = (_Float16*)carve((size_t)L * D * D * 2);
  _Float16* f1_p   = (_Float16*)carve((size_t)L * D * DFF * 2);
  _Float16* f2_p   = (_Float16*)carve((size_t)L * DFF * D * 2);
  _Float16* ecw1_p = (_Float16*)carve((size_t)EFW * D * 2);
  _Float16* ecw2_p = (_Float16*)carve((size_t)D * 64 * 2);
  _Float16* ppw1_p = (_Float16*)carve((size_t)EFW * D * 2);
  _Float16* ppw2_p = (_Float16*)carve((size_t)D * 64 * 2);

  auto cdiv = [](int a, int b) { return (a + b - 1) / b; };
  auto smemB = [](int K) { return (size_t)4 * (K / 32) * 1024; };  // NT=4

  // ---- pack all weights to WMMA fragment layout (f16) ----
  for (int l = 0; l < L; ++l) {
    int sq = D * D;
    k_pack_b<<<cdiv(sq, 256), 256, 0, stream>>>(Wq + (size_t)l * sq, wq_p + (size_t)l * sq, D, D);
    k_pack_b<<<cdiv(sq, 256), 256, 0, stream>>>(Wk + (size_t)l * sq, wk_p + (size_t)l * sq, D, D);
    k_pack_b<<<cdiv(sq, 256), 256, 0, stream>>>(Wv + (size_t)l * sq, wv_p + (size_t)l * sq, D, D);
    k_pack_b<<<cdiv(sq, 256), 256, 0, stream>>>(Wo + (size_t)l * sq, wo_p + (size_t)l * sq, D, D);
    int sf = D * DFF;
    k_pack_b<<<cdiv(sf, 256), 256, 0, stream>>>(ff_w1 + (size_t)l * sf, f1_p + (size_t)l * sf, D, DFF);
    k_pack_b<<<cdiv(sf, 256), 256, 0, stream>>>(ff_w2 + (size_t)l * sf, f2_p + (size_t)l * sf, DFF, D);
  }
  k_pack_b<<<cdiv(EFW * D, 256), 256, 0, stream>>>(ec_w1, ecw1_p, EFW, D);
  k_pack_b<<<cdiv(D * 64, 256), 256, 0, stream>>>(ec_w2, ecw2_p, D, 64);
  k_pack_b<<<cdiv(EFW * D, 256), 256, 0, stream>>>(pp_w1, ppw1_p, EFW, D);
  k_pack_b<<<cdiv(D * 64, 256), 256, 0, stream>>>(pp_w2, ppw2_p, D, 64);

  // ---- input projection + edge embedding ----
  k_node_proj<<<cdiv(N * D, 256), 256, 0, stream>>>(x, in_w, in_b, h, h16);
  k_edge_embed<<<cdiv(E, 8), 256, 0, stream>>>(ea, ee_w1, ee_b1, ee_w2, ee_b2, efeat, ew);

  // ---- transformer layers ----
  dim3 gD(cdiv(N, 128), D / 64);    // block tile 128x64, Nc=128
  dim3 gF(cdiv(N, 128), DFF / 64);  // Nc=512
  for (int l = 0; l < L; ++l) {
    size_t sq = (size_t)l * D * D;
    size_t sf = (size_t)l * D * DFF;
    // Q, K, V
    k_gemm16<4, false><<<gD, 256, smemB(D), stream>>>(h16, wq_p + sq, nullptr, Qb, nullptr, N, D, D);
    k_gemm16<4, false><<<gD, 256, smemB(D), stream>>>(h16, wk_p + sq, nullptr, Kb, nullptr, N, D, D);
    k_gemm16<4, false><<<gD, 256, smemB(D), stream>>>(h16, wv_p + sq, nullptr, Vb, nullptr, N, D, D);
    // edge scatter
    k_zero<<<cdiv(N * D, 256), 256, 0, stream>>>(sums, N * D);
    k_zero<<<cdiv(N, 256), 256, 0, stream>>>(cnt, N);
    k_attn_edge<<<cdiv(E, 8), 256, 0, stream>>>(Qb, Kb, Vb, src, dst, ew, sums, cnt);
    k_segmean<<<cdiv(N * D, 256), 256, 0, stream>>>(sums, cnt, msg16);
    // output projection + LN1
    k_gemm16<4, false><<<gD, 256, smemB(D), stream>>>(msg16, wo_p + sq, Wo_b + (size_t)l * D,
                                                      aout, nullptr, N, D, D);
    k_ln_res<<<cdiv(N, 8), 256, 0, stream>>>(h, h16, aout, ln1_s + (size_t)l * D,
                                             ln1_b + (size_t)l * D);
    // FFN + LN2
    k_gemm16<4, true><<<gF, 256, smemB(D), stream>>>(h16, f1_p + sf, ff_b1 + (size_t)l * DFF,
                                                     nullptr, hid16, N, D, DFF);
    k_gemm16<4, false><<<gD, 256, smemB(DFF), stream>>>(hid16, f2_p + sf, ff_b2 + (size_t)l * D,
                                                        aout, nullptr, N, DFF, D);
    k_ln_res<<<cdiv(N, 8), 256, 0, stream>>>(h, h16, aout, ln2_s + (size_t)l * D,
                                             ln2_b + (size_t)l * D);
  }

  // ---- edge heads, chunked over edges ----
  for (int c0 = 0; c0 < E; c0 += CH) {
    int Ec = (E - c0 < CH) ? (E - c0) : CH;
    dim3 g1(cdiv(Ec, 128), D / 64);
    dim3 g2(cdiv(Ec, 128), 1);
    // edge-classifier chain
    k_gemm_ef<<<g1, 256, smemB(EFW), stream>>>(h16, efeat, src, dst, ecw1_p, ec_b1, z1, c0, Ec, D);
    k_gemm16<4, true><<<g2, 256, smemB(D), stream>>>(z1, ecw2_p, ec_b2, nullptr, z2, Ec, D, 64);
    k_headout<1, false><<<cdiv(Ec, 256), 256, 0, stream>>>(z2, ec_w3, ec_b3, out + c0, Ec);
    // param-predictor chain
    k_gemm_ef<<<g1, 256, smemB(EFW), stream>>>(h16, efeat, src, dst, ppw1_p, pp_b1, z1, c0, Ec, D);
    k_gemm16<4, true><<<g2, 256, smemB(D), stream>>>(z1, ppw2_p, pp_b2, nullptr, z2, Ec, D, 64);
    k_headout<NEP, true><<<cdiv(Ec, 256), 256, 0, stream>>>(
        z2, pp_w3, pp_b3, out + (size_t)E + (size_t)c0 * NEP, Ec);
  }
}
